// GNN_layer_60610578482039
// MI455X (gfx1250) — matrix operand order repaired and checked by reading, not compiled, not run on hardware
//
#include <hip/hip_runtime.h>
#include <hip/hip_bf16.h>

// Problem constants (from reference)
#define NB   64    // batches
#define NN   512   // nodes
#define CIN  32
#define COUT 64
#define XNUM 68    // 4 + 2*CIN

typedef __attribute__((ext_vector_type(8)))  float  v8f;
typedef __attribute__((ext_vector_type(16))) __bf16 v16bf;
typedef __attribute__((ext_vector_type(8)))  __bf16 v8bf;

// Workspace layout (float offsets for f32 region, then bf16 region)
#define OFF_VEC  0u                                   // [NB][NN]
#define OFF_MC   (OFF_VEC + NB*NN)                    // [NB][NN] mean_cols
#define OFF_DG   (OFF_MC + NB*NN)                     // [NB][NN] diag
#define OFF_MX   (OFF_DG + NB*NN)                     // [NB][CIN] mean_X
#define OFF_SCAL (OFF_MX + NB*CIN)                    // [NB][4]: md, ma, const
#define OFF_S1   (OFF_SCAL + NB*4)                    // [NB][COUT]
#define OFF_S2   (OFF_S1 + NB*COUT)                   // [NB][COUT]
#define OFF_BF   (OFF_S2 + NB*COUT)                   // bf16 region starts here
#define X1ELEMS  ((size_t)NB * COUT * NN)             // per hi/lo array

__device__ __forceinline__ v16bf cat8(v8bf a, v8bf b) {
    return __builtin_shufflevector(a, b, 0,1,2,3,4,5,6,7,8,9,10,11,12,13,14,15);
}

// ---------------------------------------------------------------------------
// Kernel 1: per-batch statistics. grid=NB, block=512 (16 waves)
// ---------------------------------------------------------------------------
__global__ void k_stats(const float* __restrict__ A, const float* __restrict__ X,
                        const float* __restrict__ c, float* __restrict__ ws) {
    const int n = blockIdx.x;
    const float* An = A + (size_t)n * NN * NN;
    const float* Xn = X + (size_t)n * NN * CIN;

    __shared__ float s_mc[NN];
    __shared__ float s_dg[NN];
    __shared__ float s_red[NN];
    __shared__ float s_mx[CIN];
    __shared__ float s_c[96];
    __shared__ float s_md, s_ma;

    const int t = threadIdx.x;
    const int wv = t >> 5, lane = t & 31;

    if (t < 5 + 2 * CIN) s_c[t] = c[t];

    // Row means of A: wave wv handles rows wv*32..wv*32+31 (coalesced per wave)
    for (int r = 0; r < 32; ++r) {
        const int i = wv * 32 + r;
        const float* row = An + (size_t)i * NN;
        float sum = 0.f;
        #pragma unroll 4
        for (int j = lane; j < NN; j += 32) sum += row[j];
        #pragma unroll
        for (int off = 16; off > 0; off >>= 1) sum += __shfl_down(sum, off, 32);
        if (lane == 0) s_mc[i] = sum * (1.0f / NN);
    }
    s_dg[t] = An[(size_t)t * NN + t];
    __syncthreads();

    // mean_diag
    s_red[t] = s_dg[t];
    __syncthreads();
    for (int st = 256; st > 0; st >>= 1) { if (t < st) s_red[t] += s_red[t + st]; __syncthreads(); }
    if (t == 0) s_md = s_red[0] * (1.0f / NN);
    __syncthreads();
    // mean_all
    s_red[t] = s_mc[t];
    __syncthreads();
    for (int st = 256; st > 0; st >>= 1) { if (t < st) s_red[t] += s_red[t + st]; __syncthreads(); }
    if (t == 0) s_ma = s_red[0] * (1.0f / NN);
    __syncthreads();

    // mean_X[c] = mean_i X[n,i,c]
    {
        const int cc = t & 31, g = t >> 5;
        float p = 0.f;
        #pragma unroll 4
        for (int r = 0; r < 32; ++r) p += Xn[(size_t)(g * 32 + r) * CIN + cc];
        s_red[g * 32 + cc] = p;
    }
    __syncthreads();
    if (t < CIN) {
        float p = 0.f;
        #pragma unroll
        for (int g = 0; g < 16; ++g) p += s_red[g * 32 + t];
        s_mx[t] = p * (1.0f / NN);
    }
    __syncthreads();

    // vec[i] = c3*mean_cols + c4*diag + sum_c w6[c]*X[n,i,c]
    {
        float sC = 0.f;
        const float* xr = Xn + (size_t)t * CIN;
        #pragma unroll 8
        for (int cc = 0; cc < CIN; ++cc) sC += s_c[5 + cc] * xr[cc];
        const float vecv = s_c[3] * s_mc[t] + s_c[4] * s_dg[t] + sC;
        ws[OFF_VEC + n * NN + t] = vecv;
        ws[OFF_MC  + n * NN + t] = s_mc[t];
        ws[OFF_DG  + n * NN + t] = s_dg[t];
    }
    if (t < CIN) ws[OFF_MX + n * CIN + t] = s_mx[t];
    if (t == 0) {
        float cst = s_c[1] * s_ma + s_c[2] * s_md;
        #pragma unroll 8
        for (int cc = 0; cc < CIN; ++cc) cst += s_c[5 + CIN + cc] * s_mx[cc];
        ws[OFF_SCAL + n * 4 + 0] = s_md;
        ws[OFF_SCAL + n * 4 + 1] = s_ma;
        ws[OFF_SCAL + n * 4 + 2] = cst;
    }
}

// ---------------------------------------------------------------------------
// Kernel 2: X1 (bf16 hi/lo split) -> ws, X2 term -> d_out, S1/S2 (exact f32).
// grid=NB, block=256
// ---------------------------------------------------------------------------
__global__ void k_x1x2(const float* __restrict__ X,
                       const float* __restrict__ W1, const float* __restrict__ W2,
                       float* __restrict__ ws, float* __restrict__ out) {
    const int n = blockIdx.x;
    const float* Xn = X + (size_t)n * NN * CIN;
    __bf16* X1H = (__bf16*)(ws + OFF_BF);
    __bf16* X1L = X1H + X1ELEMS;

    __shared__ float W1s[COUT * XNUM];
    __shared__ float W2s[COUT * XNUM];
    __shared__ float mcs[NN], dgs[NN], vs[NN];
    __shared__ float mxs[CIN];
    __shared__ float t1s[COUT], t2s[COUT];
    __shared__ float s1p[4 * COUT], s2p[4 * COUT];
    __shared__ float s_md, s_ma;

    const int t = threadIdx.x;
    for (int idx = t; idx < COUT * XNUM; idx += 256) { W1s[idx] = W1[idx]; W2s[idx] = W2[idx]; }
    for (int idx = t; idx < NN; idx += 256) {
        mcs[idx] = ws[OFF_MC  + n * NN + idx];
        dgs[idx] = ws[OFF_DG  + n * NN + idx];
        vs[idx]  = ws[OFF_VEC + n * NN + idx];
    }
    if (t < CIN) mxs[t] = ws[OFF_MX + n * CIN + t];
    if (t == 0) { s_md = ws[OFF_SCAL + n * 4 + 0]; s_ma = ws[OFF_SCAL + n * 4 + 1]; }
    __syncthreads();

    if (t < COUT) {
        float a1 = 0.f, a2 = 0.f;
        #pragma unroll 8
        for (int cc = 0; cc < CIN; ++cc) {
            a1 += W1s[t * XNUM + CIN + cc] * mxs[cc];
            a2 += W2s[t * XNUM + CIN + cc] * mxs[cc];
        }
        t1s[t] = a1 + W1s[t * XNUM + 66] * s_md + W1s[t * XNUM + 67] * s_ma;
        t2s[t] = a2 + W2s[t * XNUM + 66] * s_md + W2s[t * XNUM + 67] * s_ma;
    }
    __syncthreads();

    const int b = t & (COUT - 1);
    const int q = t >> 6;               // j-quarter
    float s1 = 0.f, s2 = 0.f;
    const float w1mc = W1s[b * XNUM + 64], w1dg = W1s[b * XNUM + 65];
    const float w2mc = W2s[b * XNUM + 64], w2dg = W2s[b * XNUM + 65];

    for (int jj = 0; jj < NN / 4; ++jj) {
        const int j = q * (NN / 4) + jj;
        const float* xr = Xn + (size_t)j * CIN;
        float x1 = t1s[b], x2 = t2s[b];
        #pragma unroll 8
        for (int cc = 0; cc < CIN; ++cc) {
            const float xv = xr[cc];
            x1 += W1s[b * XNUM + cc] * xv;
            x2 += W2s[b * XNUM + cc] * xv;
        }
        x1 += w1mc * mcs[j] + w1dg * dgs[j];
        x2 += w2mc * mcs[j] + w2dg * dgs[j];
        // split-precision store of X1: x1 = hi + lo (lo exact residual to bf16)
        const size_t xi = ((size_t)n * COUT + b) * NN + j;
        const __bf16 h = (__bf16)x1;
        X1H[xi] = h;
        X1L[xi] = (__bf16)(x1 - (float)h);
        out[(size_t)n * NN * COUT + (size_t)j * COUT + b] = x2;   // coalesced in b
        s1 += x1;
        s2 += vs[j] * x1;
    }
    s1p[q * COUT + b] = s1;
    s2p[q * COUT + b] = s2;
    __syncthreads();
    if (t < COUT) {
        ws[OFF_S1 + n * COUT + t] = s1p[t] + s1p[COUT + t] + s1p[2 * COUT + t] + s1p[3 * COUT + t];
        ws[OFF_S2 + n * COUT + t] = s2p[t] + s2p[COUT + t] + s2p[2 * COUT + t] + s2p[3 * COUT + t];
    }
}

// ---------------------------------------------------------------------------
// Kernel 3: G = A @ X1^T via bf16x3 split with V_WMMA_F32_16X16X32_BF16.
// grid=(NN/32, NB), block=256 (8 waves = 2 i-tiles x 4 b-tiles of 16x16)
// ---------------------------------------------------------------------------
__global__ void k_gemm(const float* __restrict__ A, const float* __restrict__ Ac,
                       const float* __restrict__ ws_c, float* __restrict__ out) {
    const float* ws = ws_c;
    const __bf16* X1H = (const __bf16*)(ws + OFF_BF);
    const __bf16* X1L = X1H + X1ELEMS;
    const int i0 = blockIdx.x * 32;
    const int n  = blockIdx.y;
    const float* An = A + (size_t)n * NN * NN;

    // stride 136 (x2B = 272B rows, 16B aligned); K-chunk = 128
    __shared__ __align__(16) __bf16 Ah[32][136];
    __shared__ __align__(16) __bf16 Al[32][136];
    __shared__ __align__(16) __bf16 Bh[64][136];
    __shared__ __align__(16) __bf16 Bl[64][136];

    const int t = threadIdx.x;
    const int wv = t >> 5, lane = t & 31;
    const int isub  = wv & 1;            // i tile within block (0/1)
    const int btile = wv >> 1;           // b tile (0..3)
    const int m    = lane & 15;
    const int koff = (lane >> 4) << 3;   // A frag: K pair-group offset (0/8)
    const int boff = (lane >> 4) << 4;   // unused (0/16); B frag uses koff like A

    // cooperative-load coordinates
    const int arow = t >> 3, acol = (t & 7) * 16;   // A: 32 x 128 f32, 16/thread
    const int brow = t >> 2, bcol = (t & 3) * 32;   // B: 64 x 128 bf16, 32/thread
    const __bf16* bhsrc = X1H + ((size_t)n * COUT + brow) * NN + bcol;
    const __bf16* blsrc = X1L + ((size_t)n * COUT + brow) * NN + bcol;

    v8f acc = {};

    for (int k0 = 0; k0 < NN; k0 += 128) {
        // --- stage A: load f32, split to bf16 hi/lo in registers ---
        const float* asrc = An + (size_t)(i0 + arow) * NN + k0 + acol;
        float av[16];
        *(float4*)&av[0]  = *(const float4*)(asrc);
        *(float4*)&av[4]  = *(const float4*)(asrc + 4);
        *(float4*)&av[8]  = *(const float4*)(asrc + 8);
        *(float4*)&av[12] = *(const float4*)(asrc + 12);
        v8bf h0, h1, l0, l1;
        #pragma unroll
        for (int e = 0; e < 8; ++e) {
            const __bf16 h = (__bf16)av[e];
            h0[e] = h; l0[e] = (__bf16)(av[e] - (float)h);
            const __bf16 g = (__bf16)av[8 + e];
            h1[e] = g; l1[e] = (__bf16)(av[8 + e] - (float)g);
        }
        *(v8bf*)&Ah[arow][acol]     = h0;
        *(v8bf*)&Ah[arow][acol + 8] = h1;
        *(v8bf*)&Al[arow][acol]     = l0;
        *(v8bf*)&Al[arow][acol + 8] = l1;
        // --- stage B: bf16 hi/lo already split, straight copy ---
        #pragma unroll
        for (int u = 0; u < 32; u += 8) {
            *(v8bf*)&Bh[brow][bcol + u] = *(const v8bf*)(bhsrc + k0 + u);
            *(v8bf*)&Bl[brow][bcol + u] = *(const v8bf*)(blsrc + k0 + u);
        }
        if (k0 + 128 < NN)
            __builtin_prefetch(asrc + 128, 0, 3);     // global_prefetch next A chunk
        __syncthreads();

        const __bf16* arh = &Ah[isub * 16 + m][0];
        const __bf16* arl = &Al[isub * 16 + m][0];
        const __bf16* brh = &Bh[btile * 16 + m][0];
        const __bf16* brl = &Bl[btile * 16 + m][0];
        #pragma unroll
        for (int ks = 0; ks < 128; ks += 32) {
            // ISA layouts: A 16x32 -> [koff..koff+7] + [16+koff..23+koff];
            //              B 32x16 -> same K split as A at column (lane&15)
            v16bf ahv = cat8(*(const v8bf*)(arh + ks + koff), *(const v8bf*)(arh + ks + 16 + koff));
            v16bf alv = cat8(*(const v8bf*)(arl + ks + koff), *(const v8bf*)(arl + ks + 16 + koff));
            v16bf bhv = cat8(*(const v8bf*)(brh + ks + koff), *(const v8bf*)(brh + ks + 16 + koff));
            v16bf blv = cat8(*(const v8bf*)(brl + ks + koff), *(const v8bf*)(brl + ks + 16 + koff));
            // (hi+lo)*(hi+lo) ~= hi*hi + hi*lo + lo*hi  (lo*lo below f32 noise)
            acc = __builtin_amdgcn_wmma_f32_16x16x32_bf16(false, ahv, false, bhv, (short)0, acc, false, false);
            acc = __builtin_amdgcn_wmma_f32_16x16x32_bf16(false, ahv, false, blv, (short)0, acc, false, false);
            acc = __builtin_amdgcn_wmma_f32_16x16x32_bf16(false, alv, false, bhv, (short)0, acc, false, false);
        }
        __syncthreads();
    }

    // Epilogue: out[n,i,b] += (c0/NN)*G + (1/NN)*((const+vec[i])*S1[b] + S2[b])
    const float c0n = Ac[0] * (1.0f / NN);
    const float cst = ws[OFF_SCAL + n * 4 + 2];
    const int bb = btile * 16 + m;
    const float s1b = ws[OFF_S1 + n * COUT + bb];
    const float s2b = ws[OFF_S2 + n * COUT + bb];
    const int ihalf = (lane >> 4) << 3;
    float* outn = out + (size_t)n * NN * COUT;
    #pragma unroll
    for (int v = 0; v < 8; ++v) {
        const int i = i0 + isub * 16 + v + ihalf;
        const float vi = ws[OFF_VEC + n * NN + i];
        outn[(size_t)i * COUT + bb] +=
            c0n * acc[v] + (1.0f / NN) * ((cst + vi) * s1b + s2b);
    }
}

// ---------------------------------------------------------------------------
extern "C" void kernel_launch(void* const* d_in, const int* in_sizes, int n_in,
                              void* d_out, int out_size, void* d_ws, size_t ws_size,
                              hipStream_t stream) {
    const float* A  = (const float*)d_in[0];
    const float* X  = (const float*)d_in[1];
    const float* Ac = (const float*)d_in[2];
    const float* W1 = (const float*)d_in[3];
    const float* W2 = (const float*)d_in[4];
    float* out = (float*)d_out;
    float* ws  = (float*)d_ws;
    (void)in_sizes; (void)n_in; (void)out_size; (void)ws_size;

    k_stats<<<NB, 512, 0, stream>>>(A, X, Ac, ws);
    k_x1x2 <<<NB, 256, 0, stream>>>(X, W1, W2, ws, out);
    dim3 g3(NN / 32, NB);
    k_gemm <<<g3, 256, 0, stream>>>(A, Ac, ws, out);
}
